// HierarchicalFNO_30760555774568
// MI455X (gfx1250) — matrix operand order repaired and checked by reading, not compiled, not run on hardware
//
#include <hip/hip_runtime.h>
#include <hip/hip_bf16.h>

// ---------------------------------------------------------------------------
// HierarchicalFNO for MI455X (gfx1250, wave32, WMMA).
//
// All heavy stages are 16x16x32 f16 WMMA GEMMs with f32 accumulation:
//   conv      : implicit GEMM [16 pixels x 576] x [576 x 64]; K re-ordered as
//               (tap, cin); x kept as halo-padded NHWC f16 so each A fragment
//               is two unconditional global b128 loads at immediate offsets.
//   fwd DFT   : X1 = x(128x128) @ E(128x64) ; X2 = A2(32x256) @ [X1re;X1im]
//   mode mul  : per (u,v): complex [32x64] x [64x64]  (4 real WMMA GEMMs)
//   inv DFT   : Z = G(128x64) @ packed(64x32) ; out = [Zre|Zim] @ Bw(64x128)
// DFT matrices / spectral kernel / conv weights are pre-packed into the exact
// WMMA fragment register layout (one or two 16B loads per lane per fragment).
// LDS intermediates are stored k-contiguous (transposed + padded).
// ---------------------------------------------------------------------------

typedef __attribute__((ext_vector_type(16))) _Float16 v16h;
typedef __attribute__((ext_vector_type(8)))  _Float16 v8h;
typedef __attribute__((ext_vector_type(8)))  float    v8f;

#define TWO_PI 6.28318530717958647692f

// ISA 7.12.2 fragment coordinate maps (wave32, 16-bit elements)
__device__ __forceinline__ int a_k_of_t(int lane, int t) {
  return ((t < 8) ? t : t + 8) + ((lane & 16) ? 8 : 0);
}
__device__ __forceinline__ int b_k_of_t(int lane, int t) {
  return ((t < 8) ? t : t + 8) + ((lane & 16) ? 8 : 0);
}

__device__ __forceinline__ v8f wmma_f16(v16h a, v16h b, v8f c) {
  return __builtin_amdgcn_wmma_f32_16x16x32_f16(
      /*neg_a=*/false, a, /*neg_b=*/false, b,
      /*c_mod=*/(short)0, c, /*reuse_a=*/false, /*reuse_b=*/false);
}

__device__ __forceinline__ v16h cat16(v8h lo, v8h hi) {
  return __builtin_shufflevector(lo, hi, 0, 1, 2, 3, 4, 5, 6, 7,
                                 8, 9, 10, 11, 12, 13, 14, 15);
}

#define NB   32
#define NC   64     // CIN
#define NO   64     // COUT
#define NH   128
#define NW   128
#define NHP  130    // padded (1-px halo) for conv
#define NT   32     // kept modes per axis
#define NMODE (NT*NT)

static constexpr size_t N_X     = (size_t)NB * NC * NH * NW;       // 33,554,432
static constexpr size_t SZ_XH   = N_X * 2;                         // NCHW f16
static constexpr size_t SZ_XT   = (size_t)NB * NHP * NHP * NC * 2; // padded NHWC
static constexpr size_t SZ_WFRAG = (size_t)18 * 4 * 512 * 2;
static constexpr size_t SZ_TAB  = (size_t)8192 * 2;
static constexpr size_t SZ_KH   = (size_t)NMODE * 2 * 4 * 512 * 2;
static constexpr size_t SZ_XHAT = (size_t)NMODE * NB * NC * 2;
static constexpr size_t SZ_OF   = (size_t)NB * NO * NMODE * 2;

// ---------------------------------------------------------------------------
// K0a: DFT matrices, pre-packed in fragment layout (5 tables x 8192 halves).
// Angles are pre-reduced to [0,2pi) -> hardware v_sin/v_cos is exact enough
// for f16 twiddles and ~20x fewer VALU ops than libm sinf/cosf.
// ---------------------------------------------------------------------------
__global__ void k_fill_consts(_Float16* __restrict__ Efrag,
                              _Float16* __restrict__ A2re,
                              _Float16* __restrict__ A2im,
                              _Float16* __restrict__ Gfrag,
                              _Float16* __restrict__ Bwfrag) {
  int idx = blockIdx.x * 256 + threadIdx.x;
  if (idx >= 5 * 8192) return;
  int region = idx >> 13;
  int i = idx & 8191;
  int lane = (i >> 4) & 31;
  int t = i & 15;
  float val = 0.f;
  if (region == 0) {                       // Efrag[kc4][nt4]: k=w, n: [cos|-sin]
    int kc = i >> 11, nt = (i >> 9) & 3;
    int k = kc * 32 + b_k_of_t(lane, t);
    int n = nt * 16 + (lane & 15);
    int v = n & 31;
    float ang = TWO_PI * (float)((k * v) & 127) / 128.f;
    val = (n < 32) ? __cosf(ang) : -__sinf(ang);
    Efrag[i] = (_Float16)val;
  } else if (region == 1 || region == 2) { // A2[kc8][mt2]: m=u, k' in 0..255
    int kc = i >> 10, mt = (i >> 9) & 1;
    int u = mt * 16 + (lane & 15);
    int kp = kc * 32 + a_k_of_t(lane, t);
    int h = (kp < 128) ? kp : kp - 128;
    float ang = TWO_PI * (float)((u * h) & 127) / 128.f;
    if (region == 1) val = (kp < 128) ? __cosf(ang) : __sinf(ang);   // [Fre|-Fim]
    else             val = (kp < 128) ? -__sinf(ang) : __cosf(ang);  // [Fim| Fre]
    (region == 1 ? A2re : A2im)[i] = (_Float16)val;
  } else if (region == 3) {                // Gfrag[kc2][mt8]: m=h, [cos|sin]
    int kc = i >> 12, mt = (i >> 9) & 7;
    int h = mt * 16 + (lane & 15);
    int kp = kc * 32 + a_k_of_t(lane, t);
    int u = (kp < 32) ? kp : kp - 32;
    float ang = TWO_PI * (float)((u * h) & 127) / 128.f;
    val = (kp < 32) ? __cosf(ang) : __sinf(ang);
    Gfrag[i] = (_Float16)val;
  } else {                                 // Bwfrag[kc2][nt8]: [cos;-sin]
    int kc = i >> 12, nt = (i >> 9) & 7;
    int k = kc * 32 + b_k_of_t(lane, t);
    int w = nt * 16 + (lane & 15);
    int v = (k < 32) ? k : k - 32;
    float ang = TWO_PI * (float)((v * w) & 127) / 128.f;
    val = (k < 32) ? __cosf(ang) : -__sinf(ang);
    Bwfrag[i] = (_Float16)val;
  }
}

// K0b: conv weight -> B fragments; reduction order k = (ki*3+kj)*64 + c
__global__ void k_pack_wfrag(const float* __restrict__ weight,
                             _Float16* __restrict__ wfrag) {
  int idx = blockIdx.x * 256 + threadIdx.x;
  if (idx >= 18 * 4 * 512) return;
  int t = idx & 15, lane = (idx >> 4) & 31;
  int nt = (idx >> 9) & 3, kc = idx >> 11;
  int k = kc * 32 + b_k_of_t(lane, t);       // 0..575
  int o = nt * 16 + (lane & 15);
  int c = k & 63, tap = k >> 6;              // tap = ki*3+kj
  wfrag[idx] = (_Float16)weight[((size_t)o * NC + c) * 9 + tap];
}

// K0c: spectral kernel k_hat(u,v,c,o), packed as B fragments (k=c, n=o)
__global__ void k_khat(const float* __restrict__ weight,
                       _Float16* __restrict__ khre,
                       _Float16* __restrict__ khim) {
  int idx = blockIdx.x * 256 + threadIdx.x;
  int t = idx & 15, lane = (idx >> 4) & 31;
  int nt = (idx >> 9) & 3, kt = (idx >> 11) & 1;
  int mode = idx >> 12;
  int u = mode >> 5, vv = mode & 31;
  int c = kt * 32 + b_k_of_t(lane, t);
  int o = nt * 16 + (lane & 15);
  const float* w = &weight[((size_t)o * NC + c) * 9];
  float re = 0.f, im = 0.f;
#pragma unroll
  for (int i = 0; i < 3; ++i)
#pragma unroll
    for (int j = 0; j < 3; ++j) {
      float ang = TWO_PI * (float)((u * i + vv * j) & 127) / 128.f;
      float wv = w[i * 3 + j];
      re += wv * __cosf(ang);
      im -= wv * __sinf(ang);
    }
  khre[idx] = (_Float16)re;
  khim[idx] = (_Float16)im;
}

// K0d: zero-fill the 1-px halo ring of padded NHWC xt
__global__ void k_halo(_Float16* __restrict__ xt) {
  int idx = blockIdx.x * 256 + threadIdx.x;    // NB * 4 * NHP * NC
  if (idx >= NB * 4 * NHP * NC) return;
  int c = idx & 63;
  int p = (idx >> 6) % NHP;
  int side = ((idx >> 6) / NHP) & 3;
  int b = idx / (4 * NHP * NC);
  int h, w;
  if (side == 0)      { h = 0;       w = p; }
  else if (side == 1) { h = NHP - 1; w = p; }
  else if (side == 2) { h = p;       w = 0; }
  else                { h = p;       w = NHP - 1; }
  xt[(((size_t)b * NHP + h) * NHP + w) * NC + c] = (_Float16)0.f;
}

// ---------------------------------------------------------------------------
// K0e: x f32 -> f16 in BOTH layouts via LDS tile transpose.
//   xh : [B][C][H][W]          (DFT A-operand, coalesced row loads)
//   xt : [B][130][130][C]      (conv A-operand, halo-padded, c-contiguous)
// ---------------------------------------------------------------------------
__global__ void k_transpose(const float* __restrict__ x,
                            _Float16* __restrict__ xh,
                            _Float16* __restrict__ xt) {
  int bid = blockIdx.x;                  // B*H*4 = 16384
  int wt = bid & 3, h = (bid >> 2) & 127, b = bid >> 9;
  int w0 = wt * 32;
  __shared__ _Float16 tile[NC][33];
  int tid = threadIdx.x;
  for (int i = tid; i < 2048; i += 256) {        // read NCHW, 128B rows
    int c = i >> 5, w = i & 31;
    size_t src = (((size_t)b * NC + c) * NH + h) * NW + w0 + w;
    _Float16 hv = (_Float16)x[src];
    xh[src] = hv;
    tile[c][w] = hv;
  }
  __syncthreads();
  for (int i = tid; i < 2048; i += 256) {        // write NHWC (+1,+1), 128B rows
    int w = i >> 6, c = i & 63;
    xt[(((size_t)b * NHP + h + 1) * NHP + w0 + w + 1) * NC + c] = tile[c][w];
  }
}

// ---------------------------------------------------------------------------
// K1: 3x3 conv implicit GEMM. Block = 128 thr (4 waves), 16 pixels x 64 couts.
//     Halo-padded NHWC input -> no predication; per-lane base pointer + fully
//     unrolled taps -> every load is an immediate-offset global_load_b128.
// ---------------------------------------------------------------------------
__global__ void k_conv(const _Float16* __restrict__ xt,
                       const _Float16* __restrict__ wfrag,
                       const float* __restrict__ conv_bias,
                       float* __restrict__ out) {
  int bid = blockIdx.x;                  // B * H * (W/16) = 32768
  int wt = bid & 7;
  int h  = (bid >> 3) & 127;
  int b  = bid >> 10;
  int w0 = wt * 16;
  int tid = threadIdx.x, lane = tid & 31, wave = tid >> 5;
  int m = lane & 15;

  // pixel (h,w0+m) at padded coords (h+ki, w0+m+kj), halo origin offset +0
  const _Float16* abase =
      &xt[(((size_t)b * NHP + h) * NHP + (w0 + m)) * NC + ((lane & 16) ? 8 : 0)];
  const _Float16* wbase = &wfrag[((size_t)wave * 32 + lane) * 16];

  v8f acc = {};
#pragma unroll
  for (int kc = 0; kc < 18; ++kc) {
    int tap = kc >> 1;
    int ki = tap / 3, kj = tap - ki * 3;
    int aoff = (ki * NHP + kj) * NC + ((kc & 1) << 5);   // constant per iter
    v16h a = cat16(*(const v8h*)(abase + aoff), *(const v8h*)(abase + aoff + 16));
    v16h bf = *(const v16h*)(wbase + (size_t)kc * 4 * 512);
    acc = wmma_f16(a, bf, acc);
  }
  int n = lane & 15;
  int cout = wave * 16 + n;
  int moff = (lane & 16) ? 8 : 0;
  float bc = conv_bias[cout];
  size_t base = (((size_t)b * NO + cout) * NH + h) * NW + w0 + moff;
#pragma unroll
  for (int v = 0; v < 8; ++v) out[base + v] = acc[v] + bc;
}

// ---------------------------------------------------------------------------
// K2: truncated forward DFT per (b,c) image. Block = 256 thr (8 waves).
// ---------------------------------------------------------------------------
__global__ void k_fft_fwd(const _Float16* __restrict__ xh,
                          const _Float16* __restrict__ Efrag,
                          const _Float16* __restrict__ A2re,
                          const _Float16* __restrict__ A2im,
                          _Float16* __restrict__ xhat_re,
                          _Float16* __restrict__ xhat_im) {
  int bc = blockIdx.x;                   // b*64 + c
  int b = bc >> 6, c = bc & 63;
  int tid = threadIdx.x, lane = tid & 31, wave = tid >> 5;
  __shared__ _Float16 X1T[64][136];      // [col (re 0..31 | im 32..63)][h]
  const _Float16* img = xh + (size_t)bc * (NH * NW);

  for (int tile = wave; tile < 32; tile += 8) {   // S1: 8 mt x 4 nt
    int mt = tile >> 2, nt = tile & 3;
    v8f acc = {};
#pragma unroll
    for (int kc = 0; kc < 4; ++kc) {
      int mm = mt * 16 + (lane & 15);
      int kb = kc * 32 + ((lane & 16) ? 8 : 0);
      const _Float16* p = &img[(size_t)mm * NW + kb];
      v16h a = cat16(*(const v8h*)p, *(const v8h*)(p + 16));
      v16h bf = *(const v16h*)&Efrag[(((size_t)kc * 4 + nt) * 32 + lane) * 16];
      acc = wmma_f16(a, bf, acc);
    }
    int n = nt * 16 + (lane & 15);
    int moff = (lane & 16) ? 8 : 0;
    *(v8h*)&X1T[n][mt * 16 + moff] = __builtin_convertvector(acc, v8h);
  }
  __syncthreads();

  {                                      // S2: one job per wave
    int part = wave >> 2;                // 0: re, 1: im
    int mt = (wave >> 1) & 1, nt = wave & 1;
    const _Float16* A2 = part ? A2im : A2re;
    int vc = nt * 16 + (lane & 15);
    v8f acc = {};
#pragma unroll
    for (int kc = 0; kc < 8; ++kc) {
      v16h a = *(const v16h*)&A2[(((size_t)kc * 2 + mt) * 32 + lane) * 16];
      int kb = kc * 32 + ((lane & 16) ? 8 : 0);   // one side of the 128 split
      const _Float16* src = (kb < 128) ? &X1T[vc][kb] : &X1T[32 + vc][kb - 128];
      v16h bf = cat16(*(const v8h*)src, *(const v8h*)(src + 16));
      acc = wmma_f16(a, bf, acc);
    }
    int moff = (lane & 16) ? 8 : 0;
    _Float16* dst = part ? xhat_im : xhat_re;
#pragma unroll
    for (int v = 0; v < 8; ++v) {
      int u = mt * 16 + moff + v;
      dst[(((size_t)(u * 32 + vc)) * NB + b) * NC + c] = (_Float16)acc[v];
    }
  }
}

// ---------------------------------------------------------------------------
// K3: per-mode complex contraction: C = A(32b x 64c) * Khat(64c x 64o).
// ---------------------------------------------------------------------------
__global__ void k_modemul(const _Float16* __restrict__ xhat_re,
                          const _Float16* __restrict__ xhat_im,
                          const _Float16* __restrict__ khre,
                          const _Float16* __restrict__ khim,
                          _Float16* __restrict__ ofre,
                          _Float16* __restrict__ ofim) {
  int mode = blockIdx.x;
  int tid = threadIdx.x, lane = tid & 31, wave = tid >> 5;
  int mt = wave >> 2, nt = wave & 3;
  v8f cre = {}, cim = {};
  int m = mt * 16 + (lane & 15);
#pragma unroll
  for (int kt = 0; kt < 2; ++kt) {
    const _Float16* pre = &xhat_re[((size_t)mode * NB + m) * NC + kt * 32];
    const _Float16* pim = &xhat_im[((size_t)mode * NB + m) * NC + kt * 32];
    int kb = (lane & 16) ? 8 : 0;
    v16h are = cat16(*(const v8h*)(pre + kb), *(const v8h*)(pre + kb + 16));
    v16h aim = cat16(*(const v8h*)(pim + kb), *(const v8h*)(pim + kb + 16));
    v16h aimn = -aim;                       // C_re -= Aim@Bim
    size_t fo = ((((size_t)mode * 2 + kt) * 4 + nt) * 32 + lane) * 16;
    v16h bre = *(const v16h*)&khre[fo];
    v16h bim = *(const v16h*)&khim[fo];
    cre = wmma_f16(are, bre, cre);
    cre = wmma_f16(aimn, bim, cre);
    cim = wmma_f16(are, bim, cim);
    cim = wmma_f16(aim, bre, cim);
  }
  int o = nt * 16 + (lane & 15);
  int moff = (lane & 16) ? 8 : 0;
  int u = mode >> 5, vv = mode & 31;
#pragma unroll
  for (int v = 0; v < 8; ++v) {
    int bb = mt * 16 + moff + v;
    size_t idx = (((size_t)bb * NO + o) * NT + u) * NT + vv;
    ofre[idx] = (_Float16)cre[v];
    ofim[idx] = (_Float16)cim[v];
  }
}

// ---------------------------------------------------------------------------
// K4: inverse DFT per (b,o) + fused conv residual and scalar bias.
// ---------------------------------------------------------------------------
__global__ void k_ifft_out(const _Float16* __restrict__ ofre,
                           const _Float16* __restrict__ ofim,
                           const _Float16* __restrict__ Gfrag,
                           const _Float16* __restrict__ Bwfrag,
                           const float* __restrict__ bias,
                           float* __restrict__ out) {
  int bo = blockIdx.x;                   // b*64 + o
  int tid = threadIdx.x, lane = tid & 31, wave = tid >> 5;
  __shared__ _Float16 FhT[2][NT][40];    // [re/im][v][u], u-contiguous, padded
  __shared__ _Float16 Z[NH][72];         // [h][Zre 0..31 | Zim 32..63], padded
  const _Float16* sre = ofre + (size_t)bo * NMODE;
  const _Float16* sim = ofim + (size_t)bo * NMODE;
  for (int i = tid; i < NMODE; i += 256) {
    int u = i >> 5, v = i & 31;
    FhT[0][v][u] = sre[i];
    FhT[1][v][u] = sim[i];
  }
  __syncthreads();

  for (int job = wave; job < 32; job += 8) {      // stage A
    int part = job >> 4;                 // 0 -> Zre, 1 -> Zim
    int mt = (job >> 1) & 7, nt = job & 1;
    int n = nt * 16 + (lane & 15);
    int kb = (lane & 16) ? 8 : 0;
    v8f acc = {};
#pragma unroll
    for (int kc = 0; kc < 2; ++kc) {
      v16h a = *(const v16h*)&Gfrag[(((size_t)kc * 8 + mt) * 32 + lane) * 16];
      int srcpart = (part == 0) ? kc : 1 - kc;    // re:[Fre;-Fim] im:[Fim;Fre]
      const _Float16* sp = &FhT[srcpart][n][kb];
      v16h bf = cat16(*(const v8h*)sp, *(const v8h*)(sp + 16));
      if (part == 0 && kc == 1) bf = -bf;
      acc = wmma_f16(a, bf, acc);
    }
    int moff = (lane & 16) ? 8 : 0;
#pragma unroll
    for (int v = 0; v < 8; ++v)
      Z[mt * 16 + moff + v][part * 32 + n] = (_Float16)acc[v];
  }
  __syncthreads();

  float b0 = bias[0];
  for (int job = wave; job < 64; job += 8) {      // stage B
    int mt = job >> 3, nt = job & 7;
    int m = mt * 16 + (lane & 15);
    v8f acc = {};
#pragma unroll
    for (int kc = 0; kc < 2; ++kc) {
      int kb = kc * 32 + ((lane & 16) ? 8 : 0);
      v16h a = cat16(*(const v8h*)&Z[m][kb], *(const v8h*)&Z[m][kb + 16]);
      v16h bf = *(const v16h*)&Bwfrag[(((size_t)kc * 8 + nt) * 32 + lane) * 16];
      acc = wmma_f16(a, bf, acc);
    }
    int n = nt * 16 + (lane & 15);
    int moff = (lane & 16) ? 8 : 0;
    size_t base = ((size_t)bo * NH + mt * 16 + moff) * NW + n;
#pragma unroll
    for (int v = 0; v < 8; ++v) {
      size_t idx = base + (size_t)v * NW;
      out[idx] = out[idx] + acc[v] * (1.f / 16384.f) + b0;
    }
  }
}

// ---------------------------------------------------------------------------
extern "C" void kernel_launch(void* const* d_in, const int* in_sizes, int n_in,
                              void* d_out, int out_size, void* d_ws, size_t ws_size,
                              hipStream_t stream) {
  const float* x         = (const float*)d_in[0];
  const float* weight    = (const float*)d_in[1];
  const float* conv_bias = (const float*)d_in[2];
  const float* bias      = (const float*)d_in[3];
  float* out = (float*)d_out;
  (void)in_sizes; (void)n_in; (void)out_size; (void)ws_size;

  char* ws = (char*)d_ws;
  size_t off = 0;
  auto take = [&](size_t bytes) {
    char* p = ws + off;
    off += (bytes + 255) & ~(size_t)255;
    return p;
  };
  _Float16* xh     = (_Float16*)take(SZ_XH);      // 64 MB  NCHW f16
  _Float16* xt     = (_Float16*)take(SZ_XT);      // 66 MB  padded NHWC f16
  _Float16* wfrag  = (_Float16*)take(SZ_WFRAG);
  _Float16* Efrag  = (_Float16*)take(SZ_TAB);
  _Float16* A2re   = (_Float16*)take(SZ_TAB);
  _Float16* A2im   = (_Float16*)take(SZ_TAB);
  _Float16* Gfrag  = (_Float16*)take(SZ_TAB);
  _Float16* Bwfrag = (_Float16*)take(SZ_TAB);
  _Float16* khre   = (_Float16*)take(SZ_KH);      // 8 MB
  _Float16* khim   = (_Float16*)take(SZ_KH);      // 8 MB
  _Float16* xre    = (_Float16*)take(SZ_XHAT);    // 4 MB
  _Float16* xim    = (_Float16*)take(SZ_XHAT);    // 4 MB
  _Float16* ofre   = (_Float16*)take(SZ_OF);      // 4 MB
  _Float16* ofim   = (_Float16*)take(SZ_OF);      // 4 MB

  k_fill_consts<<<160, 256, 0, stream>>>(Efrag, A2re, A2im, Gfrag, Bwfrag);
  k_pack_wfrag <<<144, 256, 0, stream>>>(weight, wfrag);
  k_khat       <<<16384, 256, 0, stream>>>(weight, khre, khim);
  k_halo       <<<(NB * 4 * NHP * NC + 255) / 256, 256, 0, stream>>>(xt);
  k_transpose  <<<NB * NH * (NW / 32), 256, 0, stream>>>(x, xh, xt);
  k_conv       <<<NB * NH * (NW / 16), 128, 0, stream>>>(xt, wfrag, conv_bias, out);
  k_fft_fwd    <<<NB * NC, 256, 0, stream>>>(xh, Efrag, A2re, A2im, xre, xim);
  k_modemul    <<<NMODE, 256, 0, stream>>>(xre, xim, khre, khim, ofre, ofim);
  k_ifft_out   <<<NB * NO, 256, 0, stream>>>(ofre, ofim, Gfrag, Bwfrag, bias, out);
}
